// TransformerModel_21483426414806
// MI455X (gfx1250) — matrix operand order repaired and checked by reading, not compiled, not run on hardware
//
#include <hip/hip_runtime.h>
#include <hip/hip_bf16.h>

namespace {

constexpr int kVocab = 32000;
constexpr int kD  = 512;
constexpr int kH  = 8;
constexpr int kL  = 2;
constexpr int kDK = 64;     // kD / kH
constexpr int kB  = 2;
constexpr int kS  = 2048;
constexpr int kBS = kB * kS;                       // 4096 rows
constexpr long kLogitsN = (long)kBS * kVocab;      // 131,072,000
constexpr long kAttnN   = (long)kB * kH * kS * kS; // 67,108,864
constexpr float kEmbScale = 22.627416997969522f;   // sqrt(512)

typedef __attribute__((ext_vector_type(16))) _Float16 v16h;
typedef __attribute__((ext_vector_type(8)))  _Float16 v8h;
typedef __attribute__((ext_vector_type(8)))  float    v8f;

// ---------------------------------------------------------------------------
// Embedding gather + scale -> h (fp32) and h16 (f16)
// ---------------------------------------------------------------------------
__global__ __launch_bounds__(256) void embed_kernel(
    const int* __restrict__ x, const float* __restrict__ emb,
    float* __restrict__ h32, _Float16* __restrict__ h16)
{
  long t = (long)blockIdx.x * 256 + threadIdx.x;
  if (t >= (long)kBS * kD) return;
  int d  = (int)(t % kD);
  long bs = t / kD;
  int idx = x[bs];
  idx = idx < 0 ? 0 : (idx > kVocab - 1 ? kVocab - 1 : idx);
  float v = emb[(long)idx * kD + d] * kEmbScale;
  h32[t] = v;
  h16[t] = (_Float16)v;
}

// ---------------------------------------------------------------------------
// Weight convert+transpose: src fp32 (K x N, row major) -> dst f16 (N x K)
// ---------------------------------------------------------------------------
__global__ __launch_bounds__(256) void transpose_w_kernel(
    const float* __restrict__ src, _Float16* __restrict__ dst, int K, int N)
{
  long t = (long)blockIdx.x * 256 + threadIdx.x;
  if (t >= (long)K * N) return;
  int  n = (int)(t % N);
  long k = t / N;
  dst[(long)n * K + k] = (_Float16)src[t];
}

// ---------------------------------------------------------------------------
// V (BS x D, f16, head-interleaved) -> Vt (B*H, DK, S) f16
// ---------------------------------------------------------------------------
__global__ __launch_bounds__(256) void vtrans_kernel(
    const _Float16* __restrict__ v, _Float16* __restrict__ vt)
{
  long t = (long)blockIdx.x * 256 + threadIdx.x;
  if (t >= (long)kBS * kD) return;
  int  d  = (int)(t % kD);
  long bs = t / kD;
  int b  = (int)(bs / kS);
  int s  = (int)(bs % kS);
  int hh = d / kDK;
  int dk = d % kDK;
  vt[((long)(b * kH + hh) * kDK + dk) * kS + s] = v[t];
}

// ---------------------------------------------------------------------------
// WMMA fragment loaders (layouts per CDNA5 ISA 7.12.2, wave32)
//   A: 16x32 MxK f16, row-major, K contiguous
//   B: supplied as Bt = N x K row-major, K contiguous
// ---------------------------------------------------------------------------
__device__ __forceinline__ v16h load_a_f16(const _Float16* __restrict__ p)
{
  v8h lo = *(const v8h*)(p);
  v8h hi = *(const v8h*)(p + 16);
  v16h r;
#pragma unroll
  for (int i = 0; i < 8; ++i) { r[i] = lo[i]; r[i + 8] = hi[i]; }
  return r;
}

__device__ __forceinline__ v16h load_a_f32(const float* __restrict__ p)
{
  v8f lo = *(const v8f*)(p);
  v8f hi = *(const v8f*)(p + 16);
  v16h r;
#pragma unroll
  for (int i = 0; i < 8; ++i) { r[i] = (_Float16)lo[i]; r[i + 8] = (_Float16)hi[i]; }
  return r;
}

__device__ __forceinline__ v16h load_b_f16(const _Float16* __restrict__ p)
{
  v8h lo = *(const v8h*)(p);
  v8h hi = *(const v8h*)(p + 16);
  v16h r;
#pragma unroll
  for (int i = 0; i < 8; ++i) { r[i] = lo[i]; r[i + 8] = hi[i]; }
  return r;
}

// ---------------------------------------------------------------------------
// Generic batched GEMM:  C[m,n] = alpha * sum_k A[m,k] * Bt[n,k] (+ bias[n])
//   A: f16 (AF32=false) or fp32 (AF32=true), row-major M x K, leading dim lda
//   Bt: f16, row-major N x K, leading dim ldb
//   Output: fp32 (OUT16=false) or f16 (OUT16=true), leading dim ldc
//   Batch z: offsets = (z % bmod)*s?h + (z / bmod)*s?b
//
// Register-blocked: one wave computes a 32x64 macro-tile = 2 (M) x 4 (N)
// WMMA tiles -> 8 v_wmma per 32-wide K step from 2 A-frag + 4 B-frag loads.
// 8 waves (256 threads) per block. M % 32 == 0, N % 64 == 0, K % 32 == 0.
// ---------------------------------------------------------------------------
template <bool AF32, bool OUT16, bool HAS_BIAS>
__global__ __launch_bounds__(256) void gemm_wmma_kernel(
    const void* __restrict__ Av, const _Float16* __restrict__ Bt,
    const float* __restrict__ bias, void* __restrict__ Cv,
    int M, int N, int K, int lda, int ldb, long ldc, float alpha, int bmod,
    long sAh, long sAb, long sBh, long sBb, long sCh, long sCb)
{
  const int lane  = threadIdx.x & 31;
  const int wave  = threadIdx.x >> 5;
  const int ntn   = N >> 6;                       // macro-tiles along N (64 wide)
  const long ntiles = (long)(M >> 5) * ntn;       // macro-tiles total (32 tall)
  long tile = (long)blockIdx.x * 8 + wave;
  if (tile >= ntiles) return;                     // wave-uniform: EXEC all-ones
  const int tm = (int)(tile / ntn);
  const int tn = (int)(tile % ntn);

  const int batch = blockIdx.z;
  const int hh = batch % bmod;
  const int bb = batch / bmod;
  const long offA = (long)hh * sAh + (long)bb * sAb;
  const long offB = (long)hh * sBh + (long)bb * sBb;
  const long offC = (long)hh * sCh + (long)bb * sCb;

  const int mrow  = lane & 15;
  const int khalf = lane >> 4;

  // B fragment row pointers: n = tn*64 + j*16 + mrow, K offset khalf*8
  const _Float16* bp0 = Bt + offB + (long)(tn * 64 + 0  + mrow) * ldb + (khalf << 3);
  const _Float16* bp1 = bp0 + (long)16 * ldb;
  const _Float16* bp2 = bp0 + (long)32 * ldb;
  const _Float16* bp3 = bp0 + (long)48 * ldb;

  v8f acc[2][4];
#pragma unroll
  for (int i = 0; i < 2; ++i)
#pragma unroll
    for (int j = 0; j < 4; ++j) acc[i][j] = (v8f){};

  if constexpr (AF32) {
    const float* ap0 = (const float*)Av + offA + (long)(tm * 32 + mrow) * lda + (khalf << 3);
    const float* ap1 = ap0 + (long)16 * lda;
    for (int k = 0; k < K; k += 32) {
      v16h a0 = load_a_f32(ap0 + k);
      v16h a1 = load_a_f32(ap1 + k);
      v16h b0 = load_b_f16(bp0 + k);
      v16h b1 = load_b_f16(bp1 + k);
      v16h b2 = load_b_f16(bp2 + k);
      v16h b3 = load_b_f16(bp3 + k);
      acc[0][0] = __builtin_amdgcn_wmma_f32_16x16x32_f16(false, a0, false, b0, (short)0, acc[0][0], false, false);
      acc[0][1] = __builtin_amdgcn_wmma_f32_16x16x32_f16(false, a0, false, b1, (short)0, acc[0][1], false, false);
      acc[0][2] = __builtin_amdgcn_wmma_f32_16x16x32_f16(false, a0, false, b2, (short)0, acc[0][2], false, false);
      acc[0][3] = __builtin_amdgcn_wmma_f32_16x16x32_f16(false, a0, false, b3, (short)0, acc[0][3], false, false);
      acc[1][0] = __builtin_amdgcn_wmma_f32_16x16x32_f16(false, a1, false, b0, (short)0, acc[1][0], false, false);
      acc[1][1] = __builtin_amdgcn_wmma_f32_16x16x32_f16(false, a1, false, b1, (short)0, acc[1][1], false, false);
      acc[1][2] = __builtin_amdgcn_wmma_f32_16x16x32_f16(false, a1, false, b2, (short)0, acc[1][2], false, false);
      acc[1][3] = __builtin_amdgcn_wmma_f32_16x16x32_f16(false, a1, false, b3, (short)0, acc[1][3], false, false);
    }
  } else {
    const _Float16* ap0 = (const _Float16*)Av + offA + (long)(tm * 32 + mrow) * lda + (khalf << 3);
    const _Float16* ap1 = ap0 + (long)16 * lda;
    for (int k = 0; k < K; k += 32) {
      v16h a0 = load_a_f16(ap0 + k);
      v16h a1 = load_a_f16(ap1 + k);
      v16h b0 = load_b_f16(bp0 + k);
      v16h b1 = load_b_f16(bp1 + k);
      v16h b2 = load_b_f16(bp2 + k);
      v16h b3 = load_b_f16(bp3 + k);
      acc[0][0] = __builtin_amdgcn_wmma_f32_16x16x32_f16(false, a0, false, b0, (short)0, acc[0][0], false, false);
      acc[0][1] = __builtin_amdgcn_wmma_f32_16x16x32_f16(false, a0, false, b1, (short)0, acc[0][1], false, false);
      acc[0][2] = __builtin_amdgcn_wmma_f32_16x16x32_f16(false, a0, false, b2, (short)0, acc[0][2], false, false);
      acc[0][3] = __builtin_amdgcn_wmma_f32_16x16x32_f16(false, a0, false, b3, (short)0, acc[0][3], false, false);
      acc[1][0] = __builtin_amdgcn_wmma_f32_16x16x32_f16(false, a1, false, b0, (short)0, acc[1][0], false, false);
      acc[1][1] = __builtin_amdgcn_wmma_f32_16x16x32_f16(false, a1, false, b1, (short)0, acc[1][1], false, false);
      acc[1][2] = __builtin_amdgcn_wmma_f32_16x16x32_f16(false, a1, false, b2, (short)0, acc[1][2], false, false);
      acc[1][3] = __builtin_amdgcn_wmma_f32_16x16x32_f16(false, a1, false, b3, (short)0, acc[1][3], false, false);
    }
  }

  // C/D layout: VGPR r holds M = m0 + r with m0 = 16*tile_m + 8*khalf,
  // N = 16*tile_n + (lane&15)
#pragma unroll
  for (int i = 0; i < 2; ++i) {
    const int m0 = tm * 32 + i * 16 + (khalf << 3);
#pragma unroll
    for (int j = 0; j < 4; ++j) {
      const int n = tn * 64 + j * 16 + mrow;
      float bv = 0.0f;
      if constexpr (HAS_BIAS) bv = bias[n];
#pragma unroll
      for (int r = 0; r < 8; ++r) {
        const float v = acc[i][j][r] * alpha + bv;
        const long o = offC + (long)(m0 + r) * ldc + n;
        if constexpr (OUT16) ((_Float16*)Cv)[o] = (_Float16)v;
        else                 ((float*)Cv)[o]    = v;
      }
    }
  }
}

// ---------------------------------------------------------------------------
// In-place softmax over rows of length kS (one 256-thread block per row)
// ---------------------------------------------------------------------------
__global__ __launch_bounds__(256) void softmax_kernel(float* __restrict__ attn)
{
  __shared__ float buf[256];
  constexpr int VPT = kS / 256;  // 8
  float* row = attn + (long)blockIdx.x * kS;
  const int t = threadIdx.x;

  float vals[VPT];
  float m = -3.4e38f;
#pragma unroll
  for (int i = 0; i < VPT; ++i) { vals[i] = row[t + i * 256]; m = fmaxf(m, vals[i]); }
  buf[t] = m; __syncthreads();
  for (int s = 128; s > 0; s >>= 1) {
    if (t < s) buf[t] = fmaxf(buf[t], buf[t + s]);
    __syncthreads();
  }
  m = buf[0]; __syncthreads();

  float sum = 0.f;
#pragma unroll
  for (int i = 0; i < VPT; ++i) { vals[i] = expf(vals[i] - m); sum += vals[i]; }
  buf[t] = sum; __syncthreads();
  for (int s = 128; s > 0; s >>= 1) {
    if (t < s) buf[t] += buf[t + s];
    __syncthreads();
  }
  const float inv = 1.0f / buf[0];
#pragma unroll
  for (int i = 0; i < VPT; ++i) row[t + i * 256] = vals[i] * inv;
}

// ---------------------------------------------------------------------------
// Residual add + LayerNorm, writes h32 (in place) and h16 (one block per row)
// ---------------------------------------------------------------------------
__global__ __launch_bounds__(256) void add_ln_kernel(
    float* __restrict__ h32, _Float16* __restrict__ h16,
    const float* __restrict__ res,
    const float* __restrict__ g, const float* __restrict__ b)
{
  __shared__ float buf[256];
  const long base = (long)blockIdx.x * kD;
  const int t = threadIdx.x;

  float x0 = h32[base + t]       + res[base + t];
  float x1 = h32[base + t + 256] + res[base + t + 256];

  buf[t] = x0 + x1; __syncthreads();
  for (int s = 128; s > 0; s >>= 1) {
    if (t < s) buf[t] += buf[t + s];
    __syncthreads();
  }
  const float mu = buf[0] * (1.0f / kD);
  __syncthreads();

  const float d0 = x0 - mu, d1 = x1 - mu;
  buf[t] = d0 * d0 + d1 * d1; __syncthreads();
  for (int s = 128; s > 0; s >>= 1) {
    if (t < s) buf[t] += buf[t + s];
    __syncthreads();
  }
  const float rs = rsqrtf(buf[0] * (1.0f / kD) + 1e-5f);

  float y0 = d0 * rs * g[t]       + b[t];
  float y1 = d1 * rs * g[t + 256] + b[t + 256];
  h32[base + t]       = y0;  h16[base + t]       = (_Float16)y0;
  h32[base + t + 256] = y1;  h16[base + t + 256] = (_Float16)y1;
}

inline void* ws_take(char*& p, size_t bytes)
{
  void* r = p;
  p += (bytes + 255) & ~(size_t)255;
  return r;
}

} // namespace

// ---------------------------------------------------------------------------
extern "C" void kernel_launch(void* const* d_in, const int* in_sizes, int n_in,
                              void* d_out, int out_size, void* d_ws, size_t ws_size,
                              hipStream_t stream)
{
  (void)in_sizes; (void)n_in; (void)out_size; (void)ws_size;

  const int*   x     = (const int*)  d_in[0];
  const float* emb   = (const float*)d_in[1];
  const float* Wq    = (const float*)d_in[2];
  const float* bq    = (const float*)d_in[3];
  const float* Wk    = (const float*)d_in[4];
  const float* bk    = (const float*)d_in[5];
  const float* Wv    = (const float*)d_in[6];
  const float* bv    = (const float*)d_in[7];
  const float* Wo    = (const float*)d_in[8];
  const float* bo    = (const float*)d_in[9];
  const float* ln_g  = (const float*)d_in[10];
  const float* ln_b  = (const float*)d_in[11];
  const float* Wout  = (const float*)d_in[12];
  const float* bout  = (const float*)d_in[13];

  float* logits = (float*)d_out;                    // (B,S,VOCAB)
  float* attn0  = logits + kLogitsN;                // (B,H,S,S) per layer

  // Workspace carve-up
  char* wp = (char*)d_ws;
  _Float16* wT     = (_Float16*)ws_take(wp, (size_t)kL * 4 * kD * kD * 2); // q,k,v,o transposed f16
  _Float16* woutT  = (_Float16*)ws_take(wp, (size_t)kVocab * kD * 2);
  float*    h32    = (float*)   ws_take(wp, (size_t)kBS * kD * 4);
  _Float16* h16    = (_Float16*)ws_take(wp, (size_t)kBS * kD * 2);
  _Float16* q16    = (_Float16*)ws_take(wp, (size_t)kBS * kD * 2);
  _Float16* k16    = (_Float16*)ws_take(wp, (size_t)kBS * kD * 2);
  _Float16* v16    = (_Float16*)ws_take(wp, (size_t)kBS * kD * 2);
  _Float16* vt16   = (_Float16*)ws_take(wp, (size_t)kBS * kD * 2);
  _Float16* o16    = (_Float16*)ws_take(wp, (size_t)kBS * kD * 2);
  float*    aout32 = (float*)   ws_take(wp, (size_t)kBS * kD * 4);

  // ---- weight convert+transpose (fp32 KxN -> f16 NxK) ----
  {
    const long dd = (long)kD * kD;
    const int  blk = (int)((dd + 255) / 256);
    for (int l = 0; l < kL; ++l) {
      transpose_w_kernel<<<blk, 256, 0, stream>>>(Wq + l * dd, wT + (l * 4 + 0) * dd, kD, kD);
      transpose_w_kernel<<<blk, 256, 0, stream>>>(Wk + l * dd, wT + (l * 4 + 1) * dd, kD, kD);
      transpose_w_kernel<<<blk, 256, 0, stream>>>(Wv + l * dd, wT + (l * 4 + 2) * dd, kD, kD);
      transpose_w_kernel<<<blk, 256, 0, stream>>>(Wo + l * dd, wT + (l * 4 + 3) * dd, kD, kD);
    }
    const long dv = (long)kD * kVocab;
    transpose_w_kernel<<<(int)((dv + 255) / 256), 256, 0, stream>>>(Wout, woutT, kD, kVocab);
  }

  // ---- embedding ----
  {
    const long n = (long)kBS * kD;
    embed_kernel<<<(int)((n + 255) / 256), 256, 0, stream>>>(x, emb, h32, h16);
  }

  // GEMM launch helpers (macro-tile 32x64 per wave, 8 waves per block)
  auto grid_for = [](int M, int N, int batches) {
    const long tiles = (long)(M / 32) * (N / 64);
    return dim3((unsigned)((tiles + 7) / 8), 1, (unsigned)batches);
  };

  const long dd = (long)kD * kD;
  const long nElems = (long)kBS * kD;
  const int  elemBlk = (int)((nElems + 255) / 256);

  for (int l = 0; l < kL; ++l) {
    float* attn_l = attn0 + (long)l * kAttnN;

    // Q/K/V projections: (BS x D) = h16 @ W + b   -> f16
    gemm_wmma_kernel<false, true, true><<<grid_for(kBS, kD, 1), 256, 0, stream>>>(
        h16, wT + (l * 4 + 0) * dd, bq + l * kD, q16,
        kBS, kD, kD, kD, kD, kD, 1.0f, 1, 0, 0, 0, 0, 0, 0);
    gemm_wmma_kernel<false, true, true><<<grid_for(kBS, kD, 1), 256, 0, stream>>>(
        h16, wT + (l * 4 + 1) * dd, bk + l * kD, k16,
        kBS, kD, kD, kD, kD, kD, 1.0f, 1, 0, 0, 0, 0, 0, 0);
    gemm_wmma_kernel<false, true, true><<<grid_for(kBS, kD, 1), 256, 0, stream>>>(
        h16, wT + (l * 4 + 2) * dd, bv + l * kD, v16,
        kBS, kD, kD, kD, kD, kD, 1.0f, 1, 0, 0, 0, 0, 0, 0);

    // V -> Vt (B*H, DK, S)
    vtrans_kernel<<<elemBlk, 256, 0, stream>>>(v16, vt16);

    // scores = (Q @ K^T) / sqrt(DK), batched over z = b*H + h, fp32 into d_out slot
    gemm_wmma_kernel<false, false, false><<<grid_for(kS, kS, kB * kH), 256, 0, stream>>>(
        q16, k16, nullptr, attn_l,
        kS, kS, kDK, kD, kD, kS, 0.125f, kH,
        /*sAh*/ kDK, /*sAb*/ (long)kS * kD,
        /*sBh*/ kDK, /*sBb*/ (long)kS * kD,
        /*sCh*/ (long)kS * kS, /*sCb*/ (long)kH * kS * kS);

    // softmax in place (rows = B*H*S)
    softmax_kernel<<<kB * kH * kS, 256, 0, stream>>>(attn_l);

    // o = attn @ V : A fp32 from d_out, Bt = Vt, write f16 head-interleaved
    gemm_wmma_kernel<true, true, false><<<grid_for(kS, kDK, kB * kH), 256, 0, stream>>>(
        attn_l, vt16, nullptr, o16,
        kS, kDK, kS, kS, kS, kD, 1.0f, kH,
        /*sAh*/ (long)kS * kS, /*sAb*/ (long)kH * kS * kS,
        /*sBh*/ (long)kDK * kS, /*sBb*/ (long)kH * kDK * kS,
        /*sCh*/ kDK, /*sCb*/ (long)kS * kD);

    // output projection -> fp32 aout32
    gemm_wmma_kernel<false, false, true><<<grid_for(kBS, kD, 1), 256, 0, stream>>>(
        o16, wT + (l * 4 + 3) * dd, bo + l * kD, aout32,
        kBS, kD, kD, kD, kD, kD, 1.0f, 1, 0, 0, 0, 0, 0, 0);

    // h = LayerNorm(h + aout) * g + b   (updates h32, h16)
    add_ln_kernel<<<kBS, 256, 0, stream>>>(h32, h16, aout32, ln_g + l * kD, ln_b + l * kD);
  }

  // logits = h @ Wout + bout
  gemm_wmma_kernel<false, false, true><<<grid_for(kBS, kVocab, 1), 256, 0, stream>>>(
      h16, woutT, bout, logits,
      kBS, kVocab, kD, kD, kD, kVocab, 1.0f, 1, 0, 0, 0, 0, 0, 0);
}
